// MultiHeadAttention_8366596293180
// MI455X (gfx1250) — matrix operand order repaired and checked by reading, not compiled, not run on hardware
//
#include <hip/hip_runtime.h>

typedef __bf16 bf16_t;
typedef __attribute__((ext_vector_type(16))) __bf16 v16bf;
typedef __attribute__((ext_vector_type(8)))  __bf16 v8bf;
typedef __attribute__((ext_vector_type(8)))  float  v8f;
typedef __attribute__((ext_vector_type(4)))  unsigned int v4u;
typedef __attribute__((ext_vector_type(4)))  int v4i;
typedef __attribute__((ext_vector_type(8)))  int v8i;

#define B_  4
#define S_  2048
#define H_  16
#define D_  64
#define DM_ 1024
#define BS_ (B_ * S_)        // 8192 rows
#define N3_ (3 * DM_)        // 3072

// Q pre-scale: D^-0.5 * log2(e)  (softmax runs in base-2 domain)
#define QSCALE 0.18033688011112042f

// ---------------------------------------------------------------------------
// WMMA helpers (v_wmma_f32_16x16x32_bf16, wave32)
// ---------------------------------------------------------------------------
static __device__ __forceinline__ v8f wmma_bf16(v16bf a, v16bf b, v8f c) {
  return __builtin_amdgcn_wmma_f32_16x16x32_bf16(
      false, a, false, b, (short)0, c, false, false);
}

static __device__ __forceinline__ v16bf make16(v8bf lo, v8bf hi) {
  return __builtin_shufflevector(lo, hi, 0, 1, 2, 3, 4, 5, 6, 7,
                                 8, 9, 10, 11, 12, 13, 14, 15);
}

// A fragment 16x32 bf16: lane l = row (l&15); elem m -> K = (m/8)*16 + 8*(l>>4) + m%8
static __device__ __forceinline__ v16bf load_a_frag(const bf16_t* src, int ld,
                                                    int lane) {
  const int half = lane >> 4, ln = lane & 15;
  const bf16_t* p = src + (size_t)ln * ld + half * 8;
  v8bf lo = *(const v8bf*)(p);
  v8bf hi = *(const v8bf*)(p + 16);
  return make16(lo, hi);
}

// B fragment 32x16 from transposed storage src[n][k]: lane l = col (l&15);
// elem m -> K = (m/8)*16 + 8*(l>>4) + m%8
static __device__ __forceinline__ v16bf load_bt_frag(const bf16_t* src, int ld,
                                                     int lane) {
  const int half = lane >> 4, ln = lane & 15;
  const bf16_t* p = src + (size_t)ln * ld + half * 8;
  v8bf lo = *(const v8bf*)(p);
  v8bf hi = *(const v8bf*)(p + 16);
  return make16(lo, hi);
}

// ---------------------------------------------------------------------------
// TDM: stage a [rows x 1024] bf16 row-major A panel into LDS with one
// tensor_load_to_lds (D# built per CDNA5 ISA).
// 6-arg builtin form (clang-23 / therock-10.0 toolchain).
// Caller: wave-uniform, one wave only; follows with s_wait_tensorcnt.
// ---------------------------------------------------------------------------
static __device__ __forceinline__ void tdm_load_panel(const bf16_t* gsrc,
                                                      bf16_t* lds_dst,
                                                      int rows) {
  unsigned long long ga = (unsigned long long)(const void*)gsrc;
  unsigned lds = (unsigned)(unsigned long long)(void*)lds_dst;
  v4u g0;
  g0[0] = 1u;                                            // count=1
  g0[1] = lds;                                           // lds_addr
  g0[2] = (unsigned)ga;                                  // global_addr[31:0]
  g0[3] = (unsigned)((ga >> 32) & 0x01FFFFFFu) | (2u << 30);  // addr hi | type=2
  v8i g1;
  g1[0] = (int)(1u << 16);                   // wg_mask=0, data_size=1 (2B)
  g1[1] = (int)(((unsigned)DM_) << 16);      // tensor_dim0[15:0]=1024
  g1[2] = (int)(((unsigned)rows) << 16);     // dim0 hi=0 | tensor_dim1 lo=rows
  g1[3] = (int)(((unsigned)DM_) << 16);      // dim1 hi=0 | tile_dim0=1024
  g1[4] = (int)((unsigned)rows);             // tile_dim1=rows, tile_dim2=0
  g1[5] = (int)DM_;                          // tensor_dim0_stride lo32 = 1024
  g1[6] = 0;                                 // stride hi | dim1_stride lo
  g1[7] = 0;
  v4i g2 = {0, 0, 0, 0}, g3 = {0, 0, 0, 0};          // 2D tensor
  v8i g4 = {0, 0, 0, 0, 0, 0, 0, 0};                 // unused extra group
  __builtin_amdgcn_tensor_load_to_lds(g0, g1, g2, g3, g4, 0);
}

// ---------------------------------------------------------------------------
// Staging kernels
// ---------------------------------------------------------------------------
__global__ void k_f32_to_bf16(const float* __restrict__ in,
                              bf16_t* __restrict__ out, int n) {
  int i = blockIdx.x * blockDim.x + threadIdx.x;
  if (i < n) out[i] = (bf16_t)in[i];
}

// out[n][k] = in[k][n]; in is [rows=K][cols=N] f32, out is [N][K] bf16
__global__ void k_transpose_bf16(const float* __restrict__ in,
                                 bf16_t* __restrict__ out, int rows, int cols) {
  int i = blockIdx.x * blockDim.x + threadIdx.x;
  if (i >= rows * cols) return;
  int n = i / rows;
  int k = i - n * rows;
  out[i] = (bf16_t)in[(size_t)k * cols + n];
}

// ---------------------------------------------------------------------------
// QKV projection: qkv = Xb[8192,1024] @ w_qkv[1024,3072]
// Block = 8 waves; A panel (64 x 1024, 128KB) TDM-staged into LDS.
// Wave tile = 32M x 64N (2 M-frags x 4 N-frags, 8 wmma / k-step).
// Block tile = 64M x 256N. Epilogue scatters to Q(scaled)/K/V^T.
// ---------------------------------------------------------------------------
__global__ __launch_bounds__(256) void k_gemm_qkv(
    const bf16_t* __restrict__ X, const bf16_t* __restrict__ WT,
    bf16_t* __restrict__ Qs, bf16_t* __restrict__ Kb,
    bf16_t* __restrict__ Vt) {
  __shared__ bf16_t Als[64 * DM_];  // 128 KB
  const int lane = threadIdx.x & 31;
  const int wid = threadIdx.x >> 5;
  const int half = lane >> 4, ln = lane & 15;
  const int MbaseBlk = blockIdx.y * 64;
  const int Msub = (wid & 1) * 32;
  const int Nbase = blockIdx.x * 256 + (wid >> 1) * 64;

  if (wid == 0) {
    tdm_load_panel(X + (size_t)MbaseBlk * DM_, Als, 64);
    __builtin_amdgcn_s_wait_tensorcnt(0);
  }
  __syncthreads();

  v8f acc[2][4];
#pragma unroll
  for (int ms = 0; ms < 2; ++ms)
#pragma unroll
    for (int nt = 0; nt < 4; ++nt) acc[ms][nt] = v8f{};

  for (int kb = 0; kb < DM_; kb += 32) {
    v16bf a0 = load_a_frag(Als + (size_t)(Msub + 0) * DM_ + kb, DM_, lane);
    v16bf a1 = load_a_frag(Als + (size_t)(Msub + 16) * DM_ + kb, DM_, lane);
#pragma unroll
    for (int nt = 0; nt < 4; ++nt) {
      v16bf bw = load_bt_frag(WT + (size_t)(Nbase + nt * 16) * DM_ + kb, DM_,
                              lane);
      acc[0][nt] = wmma_bf16(a0, bw, acc[0][nt]);
      acc[1][nt] = wmma_bf16(a1, bw, acc[1][nt]);
    }
  }

#pragma unroll
  for (int ms = 0; ms < 2; ++ms) {
#pragma unroll
    for (int nt = 0; nt < 4; ++nt) {
#pragma unroll
      for (int r = 0; r < 8; ++r) {
        int rm = MbaseBlk + Msub + ms * 16 + r + 8 * half;  // [0, B*S)
        int n = Nbase + nt * 16 + ln;                       // [0, 3*DM)
        int bb = rm >> 11, s = rm & (S_ - 1);
        int which = n >> 10, rem = n & (DM_ - 1);
        int hh = rem >> 6, d = rem & (D_ - 1);
        float v = acc[ms][nt][r];
        size_t bhx = (size_t)(bb * H_ + hh);
        if (which == 0)
          Qs[(bhx * S_ + s) * D_ + d] = (bf16_t)(v * QSCALE);
        else if (which == 1)
          Kb[(bhx * S_ + s) * D_ + d] = (bf16_t)v;
        else
          Vt[(bhx * D_ + d) * S_ + s] = (bf16_t)v;  // V transposed
      }
    }
  }
}

// ---------------------------------------------------------------------------
// Flash attention (causal, base-2 softmax). 1 wave = 32 queries (2 q-subtiles)
// so K/V fragments are shared: 16 wmma per 32-key block.
// Block = 4 waves; grid = (S/32/4, H, B).
// ---------------------------------------------------------------------------
__global__ __launch_bounds__(128) void k_attention(
    const bf16_t* __restrict__ Q, const bf16_t* __restrict__ K,
    const bf16_t* __restrict__ Vt, bf16_t* __restrict__ O) {
  __shared__ bf16_t Pbuf[4][2][16][32];

  const int lane = threadIdx.x & 31;
  const int wid = threadIdx.x >> 5;
  const int half = lane >> 4, ln = lane & 15;
  const int h = blockIdx.y, b = blockIdx.z;
  const int qg = blockIdx.x * 4 + wid;  // 32-query group, 0..63
  const int q0 = qg * 32;

  const size_t bh = (size_t)(b * H_ + h);
  const bf16_t* Qp = Q + bh * S_ * D_;
  const bf16_t* Kp = K + bh * S_ * D_;
  const bf16_t* Vp = Vt + bh * D_ * S_;

  v16bf aQ[2][2];
#pragma unroll
  for (int qs = 0; qs < 2; ++qs) {
    aQ[qs][0] = load_a_frag(Qp + (size_t)(q0 + 16 * qs) * D_ + 0, D_, lane);
    aQ[qs][1] = load_a_frag(Qp + (size_t)(q0 + 16 * qs) * D_ + 32, D_, lane);
  }

  v8f o[2][4];
  float mr[2][8], lr[2][8];
#pragma unroll
  for (int qs = 0; qs < 2; ++qs) {
#pragma unroll
    for (int t = 0; t < 4; ++t) o[qs][t] = v8f{};
#pragma unroll
    for (int r = 0; r < 8; ++r) {
      mr[qs][r] = -1e30f;
      lr[qs][r] = 0.0f;
    }
  }

  const int nkb = qg + 1;  // keys 0 .. q0+31
  for (int kb = 0; kb < nkb; ++kb) {
    const int k0 = kb * 32;

    // K^T fragments (shared by both q-subtiles)
    v16bf b00 = load_bt_frag(Kp + (size_t)k0 * D_ + 0, D_, lane);
    v16bf b01 = load_bt_frag(Kp + (size_t)k0 * D_ + 32, D_, lane);
    v16bf b10 = load_bt_frag(Kp + (size_t)(k0 + 16) * D_ + 0, D_, lane);
    v16bf b11 = load_bt_frag(Kp + (size_t)(k0 + 16) * D_ + 32, D_, lane);

#pragma unroll
    for (int qs = 0; qs < 2; ++qs) {
      const int q0s = q0 + 16 * qs;
      v8f c0 = {}, c1 = {};
      c0 = wmma_bf16(aQ[qs][0], b00, c0);
      c0 = wmma_bf16(aQ[qs][1], b01, c0);
      c1 = wmma_bf16(aQ[qs][0], b10, c1);
      c1 = wmma_bf16(aQ[qs][1], b11, c1);

      if (k0 + 31 > q0s) {  // only blocks touching the diagonal
#pragma unroll
        for (int r = 0; r < 8; ++r) {
          int qglob = q0s + r + 8 * half;
          if (k0 + ln > qglob) c0[r] = -1e30f;
          if (k0 + 16 + ln > qglob) c1[r] = -1e30f;
        }
      }

      // online softmax in base-2 (Q pre-scaled by log2 e)
#pragma unroll
      for (int r = 0; r < 8; ++r) {
        float mx = fmaxf(c0[r], c1[r]);
#pragma unroll
        for (int s = 8; s >= 1; s >>= 1) mx = fmaxf(mx, __shfl_xor(mx, s, 32));
        float mnew = fmaxf(mr[qs][r], mx);
        float sc = exp2f(mr[qs][r] - mnew);
        float p0 = exp2f(c0[r] - mnew);
        float p1 = exp2f(c1[r] - mnew);
        c0[r] = p0;
        c1[r] = p1;
        float rs = p0 + p1;
#pragma unroll
        for (int s = 8; s >= 1; s >>= 1) rs += __shfl_xor(rs, s, 32);
        lr[qs][r] = lr[qs][r] * sc + rs;
        mr[qs][r] = mnew;
        o[qs][0][r] *= sc;
        o[qs][1][r] *= sc;
        o[qs][2][r] *= sc;
        o[qs][3][r] *= sc;
      }

      // P: C-fragment -> A-fragment relayout through LDS
#pragma unroll
      for (int r = 0; r < 8; ++r) {
        Pbuf[wid][qs][r + 8 * half][ln] = (bf16_t)c0[r];
        Pbuf[wid][qs][r + 8 * half][16 + ln] = (bf16_t)c1[r];
      }
    }

    asm volatile("s_wait_dscnt 0" ::: "memory");

    // V^T fragments (shared by both q-subtiles)
    v16bf bv0 = load_bt_frag(Vp + (size_t)(0 * 16) * S_ + k0, S_, lane);
    v16bf bv1 = load_bt_frag(Vp + (size_t)(1 * 16) * S_ + k0, S_, lane);
    v16bf bv2 = load_bt_frag(Vp + (size_t)(2 * 16) * S_ + k0, S_, lane);
    v16bf bv3 = load_bt_frag(Vp + (size_t)(3 * 16) * S_ + k0, S_, lane);

#pragma unroll
    for (int qs = 0; qs < 2; ++qs) {
      v16bf aP = load_a_frag(&Pbuf[wid][qs][0][0], 32, lane);
      o[qs][0] = wmma_bf16(aP, bv0, o[qs][0]);
      o[qs][1] = wmma_bf16(aP, bv1, o[qs][1]);
      o[qs][2] = wmma_bf16(aP, bv2, o[qs][2]);
      o[qs][3] = wmma_bf16(aP, bv3, o[qs][3]);
    }
  }

  // normalize + store to [B, S, H*D] bf16
#pragma unroll
  for (int qs = 0; qs < 2; ++qs) {
#pragma unroll
    for (int r = 0; r < 8; ++r) {
      float inv = 1.0f / lr[qs][r];
      int q = q0 + 16 * qs + r + 8 * half;
      size_t base = ((size_t)b * S_ + q) * DM_ + h * D_;
      O[base + 0 * 16 + ln] = (bf16_t)(o[qs][0][r] * inv);
      O[base + 1 * 16 + ln] = (bf16_t)(o[qs][1][r] * inv);
      O[base + 2 * 16 + ln] = (bf16_t)(o[qs][2][r] * inv);
      O[base + 3 * 16 + ln] = (bf16_t)(o[qs][3][r] * inv);
    }
  }
}

// ---------------------------------------------------------------------------
// Output projection: out[8192,1024] f32 = attnO @ w_out; same TDM-staged GEMM.
// ---------------------------------------------------------------------------
__global__ __launch_bounds__(256) void k_gemm_out(
    const bf16_t* __restrict__ A, const bf16_t* __restrict__ WT,
    float* __restrict__ out) {
  __shared__ bf16_t Als[64 * DM_];  // 128 KB
  const int lane = threadIdx.x & 31;
  const int wid = threadIdx.x >> 5;
  const int half = lane >> 4, ln = lane & 15;
  const int MbaseBlk = blockIdx.y * 64;
  const int Msub = (wid & 1) * 32;
  const int Nbase = blockIdx.x * 256 + (wid >> 1) * 64;

  if (wid == 0) {
    tdm_load_panel(A + (size_t)MbaseBlk * DM_, Als, 64);
    __builtin_amdgcn_s_wait_tensorcnt(0);
  }
  __syncthreads();

  v8f acc[2][4];
#pragma unroll
  for (int ms = 0; ms < 2; ++ms)
#pragma unroll
    for (int nt = 0; nt < 4; ++nt) acc[ms][nt] = v8f{};

  for (int kb = 0; kb < DM_; kb += 32) {
    v16bf a0 = load_a_frag(Als + (size_t)(Msub + 0) * DM_ + kb, DM_, lane);
    v16bf a1 = load_a_frag(Als + (size_t)(Msub + 16) * DM_ + kb, DM_, lane);
#pragma unroll
    for (int nt = 0; nt < 4; ++nt) {
      v16bf bw = load_bt_frag(WT + (size_t)(Nbase + nt * 16) * DM_ + kb, DM_,
                              lane);
      acc[0][nt] = wmma_bf16(a0, bw, acc[0][nt]);
      acc[1][nt] = wmma_bf16(a1, bw, acc[1][nt]);
    }
  }

#pragma unroll
  for (int ms = 0; ms < 2; ++ms)
#pragma unroll
    for (int nt = 0; nt < 4; ++nt)
#pragma unroll
      for (int r = 0; r < 8; ++r) {
        int rm = MbaseBlk + Msub + ms * 16 + r + 8 * half;
        int n = Nbase + nt * 16 + ln;
        out[(size_t)rm * DM_ + n] = acc[ms][nt][r];
      }
}

// ---------------------------------------------------------------------------
// Launcher
// ---------------------------------------------------------------------------
extern "C" void kernel_launch(void* const* d_in, const int* in_sizes, int n_in,
                              void* d_out, int out_size, void* d_ws,
                              size_t ws_size, hipStream_t stream) {
  (void)in_sizes; (void)n_in; (void)out_size; (void)ws_size;
  const float* x = (const float*)d_in[0];      // [B,S,DM]
  const float* w_qkv = (const float*)d_in[1];  // [DM, 3*DM]
  const float* w_out = (const float*)d_in[2];  // [DM, DM]
  float* out = (float*)d_out;                  // [B,S,DM] f32

  char* ws = (char*)d_ws;
  size_t off = 0;
  auto carve = [&](size_t elems) {
    bf16_t* p = (bf16_t*)(ws + off);
    off = (off + elems * sizeof(bf16_t) + 255) & ~(size_t)255;
    return p;
  };
  bf16_t* Xb    = carve((size_t)BS_ * DM_);  // x in bf16
  bf16_t* WqkvT = carve((size_t)N3_ * DM_);  // w_qkv^T bf16 [3072][1024]
  bf16_t* WoT   = carve((size_t)DM_ * DM_);  // w_out^T bf16 [1024][1024]
  bf16_t* Qs    = carve((size_t)BS_ * DM_);  // [B,H,S,D] pre-scaled
  bf16_t* Kb    = carve((size_t)BS_ * DM_);  // [B,H,S,D]
  bf16_t* Vt    = carve((size_t)BS_ * DM_);  // [B,H,D,S]
  bf16_t* AO    = carve((size_t)BS_ * DM_);  // attention out [B,S,DM]

  {
    int n = BS_ * DM_;
    k_f32_to_bf16<<<(n + 255) / 256, 256, 0, stream>>>(x, Xb, n);
    int nq = DM_ * N3_;
    k_transpose_bf16<<<(nq + 255) / 256, 256, 0, stream>>>(w_qkv, WqkvT, DM_,
                                                           N3_);
    int no = DM_ * DM_;
    k_transpose_bf16<<<(no + 255) / 256, 256, 0, stream>>>(w_out, WoT, DM_,
                                                           DM_);
  }

  // QKV projection: grid (N3/256, BS/64)
  k_gemm_qkv<<<dim3(N3_ / 256, BS_ / 64, 1), 256, 0, stream>>>(Xb, WqkvT, Qs,
                                                               Kb, Vt);

  // causal flash attention: 4 waves/block, 32 queries/wave
  k_attention<<<dim3(S_ / 32 / 4, H_, B_), 128, 0, stream>>>(Qs, Kb, Vt, AO);

  // output projection: grid (DM/256, BS/64)
  k_gemm_out<<<dim3(DM_ / 256, BS_ / 64, 1), 256, 0, stream>>>(AO, WoT, out);
}
